// Patcher_9998683865088
// MI455X (gfx1250) — compile-verified, matrix operand order checked by reading, not run on hardware
//
#include <hip/hip_runtime.h>
#include <hip/hip_bf16.h>
#include <math.h>

// ---------------------------------------------------------------------------
// Types for CDNA5 WMMA (wave32): D(f32 16x16) = A(bf16 16x32) x B(bf16 32x16) + C
// ---------------------------------------------------------------------------
typedef __attribute__((ext_vector_type(16))) __bf16         v16bf;
typedef __attribute__((ext_vector_type(16))) unsigned short v16us;
typedef __attribute__((ext_vector_type(8)))  unsigned short v8us;
typedef __attribute__((ext_vector_type(8)))  float          v8f;
typedef __attribute__((ext_vector_type(4)))  float          v4f;

#define TILE_M 128
#define TILE_N 128
#define TILE_K 32

static constexpr int B_SZ      = 4;
static constexpr int OUTER_SEQ = 2048;
static constexpr int INNER_SEQ = 512;
static constexpr int OUTER_DIM = 768;
static constexpr int INNER_DIM = 1024;
static constexpr int NHEAD     = 4;
static constexpr int NCODES    = 128;

// ---------------------------------------------------------------------------
// Device helpers
// ---------------------------------------------------------------------------
__device__ __forceinline__ unsigned short f2bf(float f) {
  // round-to-nearest-even f32 -> bf16
  unsigned int u = __float_as_uint(f);
  u += 0x7FFFu + ((u >> 16) & 1u);
  return (unsigned short)(u >> 16);
}

__device__ __forceinline__ float gelu_f(float x) {
  float x3 = x * x * x;
  return 0.5f * x * (1.0f + tanhf(0.7978845608028654f * (x + 0.044715f * x3)));
}

// Fragment load: 32 contiguous bytes per lane (two aligned ds_load_b128).
__device__ __forceinline__ v16bf load_frag(const unsigned short* p) {
  v8us lo = *(const v8us*)p;
  v8us hi = *(const v8us*)(p + 8);
  v16us u = __builtin_shufflevector(lo, hi, 0, 1, 2, 3, 4, 5, 6, 7,
                                    8, 9, 10, 11, 12, 13, 14, 15);
  return __builtin_bit_cast(v16bf, u);
}

// ---------------------------------------------------------------------------
// Generic GEMM: C[M,N] = act(alpha * A[M,K] x B + bias) + resid
// A row-major (lda); B row-major [K,N] (ldb), or TB -> B stored [N,K].
// f32 in memory, bf16 fragment-major tiles in LDS, v_wmma_f32_16x16x32_bf16.
// Block = 256 threads = 8 waves; tile 128x128; wave grid 4x2, each wave owns a
// 32x64 subtile (c[2][4] accumulators, 8 WMMA per K-step, 6 fragment loads ->
// 1.5 ds_load_b128 per WMMA).
// LDS layout (per 16x32 sub-fragment): [lane 0..31][elem 0..15] bf16, i.e. the
// exact WMMA VGPR order (ISA 7.12.2), padded to 24 elems/lane (48B, 16B-aligned)
// so fragment reads are two ds_load_b128 per operand.
// Double-buffered K pipeline: global loads of tile t+1 are issued before the
// WMMA work of tile t; LDS stores after; one barrier per K-step.
// AL=true: M%128==0, N%128==0, K%32==0, lda%4==0 (and ldb%4==0 if TB) -> all
// bounds checks fold away and A/B row loads are float4 vector loads.
// ---------------------------------------------------------------------------
template <bool AL, bool TB>
__global__ void __launch_bounds__(256) gemm_wmma_kernel(
    const float* __restrict__ A, int lda,
    const float* __restrict__ B, int ldb,
    const float* __restrict__ bias,
    const float* __restrict__ resid, int ldr,
    float* __restrict__ C, int ldc,
    int M, int N, int K, float alpha, int act) {
  // [buf][row/col subtile][lane][elem(+pad)]
  __shared__ __align__(16) unsigned short AsF[2][8][32][24];
  __shared__ __align__(16) unsigned short BsF[2][8][32][24];

  const int tid  = threadIdx.x;
  const int lane = tid & 31;
  const int wid  = tid >> 5;
  const int half = lane >> 4;
  const int l16  = lane & 15;
  const int wrow = wid & 3;          // 4 row-waves: rows 32*wrow
  const int wcol = wid >> 2;         // 2 col-waves: cols 64*wcol
  const int frow0 = wrow * 2;        // A subtile base for this wave
  const int fcol0 = wcol * 4;        // B subtile base for this wave
  const int m0   = blockIdx.y * TILE_M;
  const int n0   = blockIdx.x * TILE_N;

  // --- loader task decomposition (fixed per thread) ---
  // A: 512 tasks = 128 rows x 4 k-groups of 8; 2 tasks per thread (rows r, r+64)
  const int ar     = tid >> 2;       // 0..63
  const int ag     = tid & 3;
  const int a_st0  = ar >> 4;        // 0..3  (rows 0..63)
  const int a_st1  = a_st0 + 4;      // 4..7  (rows 64..127)
  const int a_lane = (ag & 1) * 16 + (ar & 15);
  const int a_eb   = (ag >> 1) * 8;
  // B: 512 tasks = 128 cols x 4 k-groups; 2 tasks per thread
  const int bn     = tid & 127;
  const int bg0    = tid >> 7;       // 0 or 1
  const int bg1    = bg0 + 2;        // 2 or 3
  const int b_sc   = bn >> 4;
  const int b_l16  = bn & 15;

  float ra0[8], ra1[8], rb0[8], rb1[8];

  auto ldA1 = [&](int gr, int gk, float* ra) {
    if constexpr (AL) {
      const v4f* p = (const v4f*)(A + (size_t)gr * lda + gk);
      v4f u = p[0], v = p[1];
#pragma unroll
      for (int i = 0; i < 4; ++i) { ra[i] = u[i]; ra[4 + i] = v[i]; }
    } else {
      if (gr < M && gk + 8 <= K) {
        const float* p = A + (size_t)gr * lda + gk;
#pragma unroll
        for (int i = 0; i < 8; ++i) ra[i] = p[i];
      } else {
#pragma unroll
        for (int i = 0; i < 8; ++i)
          ra[i] = (gr < M && gk + i < K) ? A[(size_t)gr * lda + gk + i] : 0.0f;
      }
    }
  };
  auto ldA = [&](int k0) {
    int gk = k0 + ag * 8;
    ldA1(m0 + ar, gk, ra0);
    ldA1(m0 + ar + 64, gk, ra1);
  };
  auto ldB = [&](int k0, int g, float* rb) {
    int gn = n0 + bn;
    int gk = k0 + g * 8;
    if constexpr (TB) {
      if constexpr (AL) {
        const v4f* p = (const v4f*)(B + (size_t)gn * ldb + gk);
        v4f u = p[0], v = p[1];
#pragma unroll
        for (int i = 0; i < 4; ++i) { rb[i] = u[i]; rb[4 + i] = v[i]; }
      } else {
        if (gn < N && gk + 8 <= K) {
          const float* p = B + (size_t)gn * ldb + gk;
#pragma unroll
          for (int i = 0; i < 8; ++i) rb[i] = p[i];
        } else {
#pragma unroll
          for (int i = 0; i < 8; ++i)
            rb[i] = (gn < N && gk + i < K) ? B[(size_t)gn * ldb + gk + i] : 0.0f;
        }
      }
    } else {
      if constexpr (AL) {
#pragma unroll
        for (int i = 0; i < 8; ++i) rb[i] = B[(size_t)(gk + i) * ldb + gn];
      } else {
#pragma unroll
        for (int i = 0; i < 8; ++i)
          rb[i] = (gn < N && gk + i < K) ? B[(size_t)(gk + i) * ldb + gn] : 0.0f;
      }
    }
  };
  auto stA = [&](int buf) {
    v8us v0, v1;
#pragma unroll
    for (int i = 0; i < 8; ++i) { v0[i] = f2bf(ra0[i]); v1[i] = f2bf(ra1[i]); }
    *(v8us*)&AsF[buf][a_st0][a_lane][a_eb] = v0;
    *(v8us*)&AsF[buf][a_st1][a_lane][a_eb] = v1;
  };
  auto stB = [&](int buf, int g, const float* rb) {
    v8us v;
#pragma unroll
    for (int i = 0; i < 8; ++i) v[i] = f2bf(rb[i]);
    *(v8us*)&BsF[buf][b_sc][(g & 1) * 16 + b_l16][(g >> 1) * 8] = v;
  };

  v8f acc[2][4];
#pragma unroll
  for (int i = 0; i < 2; ++i)
#pragma unroll
    for (int j = 0; j < 4; ++j) acc[i][j] = {};

  const int nk = (K + TILE_K - 1) / TILE_K;

  // prologue: fill buffer 0
  ldA(0);
  ldB(0, bg0, rb0);
  ldB(0, bg1, rb1);
  stA(0);
  stB(0, bg0, rb0);
  stB(0, bg1, rb1);
  __syncthreads();

  for (int t = 0; t < nk; ++t) {
    const int cur = t & 1;
    const int nxt = cur ^ 1;
    const bool more = (t + 1) < nk;

    // issue global loads for the next tile (overlap with WMMA below)
    if (more) {
      int k1 = (t + 1) * TILE_K;
      ldA(k1);
      ldB(k1, bg0, rb0);
      ldB(k1, bg1, rb1);
    }

    // compute on current tile: 32x64 per wave = 8 WMMAs from 6 fragments
    v16bf a0 = load_frag(&AsF[cur][frow0][lane][0]);
    v16bf a1 = load_frag(&AsF[cur][frow0 + 1][lane][0]);
#pragma unroll
    for (int j = 0; j < 4; ++j) {
      v16bf bj = load_frag(&BsF[cur][fcol0 + j][lane][0]);
      acc[0][j] = __builtin_amdgcn_wmma_f32_16x16x32_bf16(false, a0, false, bj, (short)0, acc[0][j], false, false);
      acc[1][j] = __builtin_amdgcn_wmma_f32_16x16x32_bf16(false, a1, false, bj, (short)0, acc[1][j], false, false);
    }

    // commit next tile to LDS
    if (more) {
      stA(nxt);
      stB(nxt, bg0, rb0);
      stB(nxt, bg1, rb1);
    }
    __syncthreads();
  }

  // epilogue
  auto store_tile = [&](v8f c, int tm, int tn) {
#pragma unroll
    for (int r = 0; r < 8; ++r) {
      int row = tm + r + 8 * half;
      int col = tn + l16;
      if (AL || (row < M && col < N)) {
        float v = alpha * c[r];
        if (bias) v += bias[col];
        if (act == 1) v = gelu_f(v);
        else if (act == 2) v = fmaxf(v, 0.0f);
        if (resid) v += resid[(size_t)row * ldr + col];
        C[(size_t)row * ldc + col] = v;
      }
    }
  };
#pragma unroll
  for (int i = 0; i < 2; ++i)
#pragma unroll
    for (int j = 0; j < 4; ++j)
      store_tile(acc[i][j], m0 + wrow * 32 + i * 16, n0 + wcol * 64 + j * 16);
}

// ---------------------------------------------------------------------------
// LayerNorm (gain only, eps=1e-5): one block per row
// ---------------------------------------------------------------------------
__global__ void __launch_bounds__(256) ln_kernel(
    const float* __restrict__ x, const float* __restrict__ g,
    float* __restrict__ y, int D) {
  __shared__ float s1[256], s2[256];
  const int tid = threadIdx.x;
  const size_t row = blockIdx.x;
  const float* xr = x + row * (size_t)D;
  float a = 0.f, b = 0.f;
  for (int i = tid; i < D; i += 256) { float v = xr[i]; a += v; b += v * v; }
  s1[tid] = a; s2[tid] = b;
  __syncthreads();
  for (int o = 128; o > 0; o >>= 1) {
    if (tid < o) { s1[tid] += s1[tid + o]; s2[tid] += s2[tid + o]; }
    __syncthreads();
  }
  float m  = s1[0] / D;
  float vv = s2[0] / D - m * m;
  float rs = rsqrtf(vv + 1e-5f);
  float* yr = y + row * (size_t)D;
  for (int i = tid; i < D; i += 256) yr[i] = (xr[i] - m) * rs * g[i];
}

// ---------------------------------------------------------------------------
// Row softmax with optional causal mask (square attention): one block per row
// Masked tail is written as exact zeros so the AV GEMM is correct.
// ---------------------------------------------------------------------------
__global__ void __launch_bounds__(256) softmax_kernel(float* __restrict__ s, int T, int causal) {
  __shared__ float red[256];
  const int tid = threadIdx.x;
  const int row = blockIdx.x;
  float* r = s + (size_t)row * T;
  const int lim = causal ? (row + 1) : T;
  float mx = -3.402823466e38f;
  for (int i = tid; i < lim; i += 256) mx = fmaxf(mx, r[i]);
  red[tid] = mx; __syncthreads();
  for (int o = 128; o > 0; o >>= 1) { if (tid < o) red[tid] = fmaxf(red[tid], red[tid + o]); __syncthreads(); }
  mx = red[0]; __syncthreads();
  float sum = 0.f;
  for (int i = tid; i < lim; i += 256) { float e = __expf(r[i] - mx); r[i] = e; sum += e; }
  red[tid] = sum; __syncthreads();
  for (int o = 128; o > 0; o >>= 1) { if (tid < o) red[tid] += red[tid + o]; __syncthreads(); }
  float inv = 1.0f / red[0];
  for (int i = tid; i < lim; i += 256) r[i] *= inv;
  for (int i = lim + tid; i < T; i += 256) r[i] = 0.f;
}

// ---------------------------------------------------------------------------
// Misc elementwise / data-movement kernels
// ---------------------------------------------------------------------------
__global__ void fill_kernel(float* __restrict__ p, float v, long long n) {
  long long i = (long long)blockIdx.x * blockDim.x + threadIdx.x;
  long long stride = (long long)gridDim.x * blockDim.x;
  for (; i < n; i += stride) p[i] = v;
}

__global__ void rowbias_kernel(const float* __restrict__ w, const float* __restrict__ b,
                               float* __restrict__ o, long long n, int D) {
  long long i = (long long)blockIdx.x * blockDim.x + threadIdx.x;
  long long stride = (long long)gridDim.x * blockDim.x;
  for (; i < n; i += stride) o[i] = w[i] + b[i % D];
}

// out[b,t,d] = rowdata[t,d] (+ x[b,t,d] if hasx)
__global__ void bcast_add_kernel(const float* __restrict__ rowdata, const float* __restrict__ x,
                                 float* __restrict__ out, long long n, long long TD, int hasx) {
  long long i = (long long)blockIdx.x * blockDim.x + threadIdx.x;
  long long stride = (long long)gridDim.x * blockDim.x;
  for (; i < n; i += stride) {
    float v = rowdata[i % TD];
    if (hasx) v += x[i];
    out[i] = v;
  }
}

// Rank-based deterministic top-k (matches jnp.top_k tie-break: earlier index wins)
__global__ void topk_kernel(const float* __restrict__ gate, int* __restrict__ idx,
                            int T, int Ksel) {
  int b = blockIdx.y;
  int i = blockIdx.x * blockDim.x + threadIdx.x;
  if (i >= T) return;
  const float* g = gate + (size_t)b * T;
  float v = g[i];
  int rank = 0;
  for (int j = 0; j < T; ++j) {
    float u = g[j];
    rank += (u > v) || (u == v && j < i);
  }
  if (rank < Ksel) idx[(size_t)b * Ksel + rank] = i;
}

__global__ void gather_kernel(const float* __restrict__ in, const int* __restrict__ idx,
                              float* __restrict__ out, int T, int Ksel, int D) {
  int bs = blockIdx.x;
  int b = bs / Ksel, s = bs % Ksel;
  int src = idx[(size_t)b * Ksel + s];
  const float* ip = in + ((size_t)b * T + src) * D;
  float* op = out + ((size_t)b * Ksel + s) * D;
  for (int c = threadIdx.x; c < D; c += 256) op[c] = ip[c];
}

__global__ void scatter_kernel(const float* __restrict__ in, const int* __restrict__ idx,
                               float* __restrict__ out, int T, int Ksel, int D) {
  int bs = blockIdx.x;
  int b = bs / Ksel, s = bs % Ksel;
  int dst = idx[(size_t)b * Ksel + s];
  const float* ip = in + ((size_t)b * Ksel + s) * D;
  float* op = out + ((size_t)b * T + dst) * D;
  for (int c = threadIdx.x; c < D; c += 256) op[c] = ip[c];
}

__global__ void binarize_kernel(float* __restrict__ z, long long n) {
  long long i = (long long)blockIdx.x * blockDim.x + threadIdx.x;
  long long stride = (long long)gridDim.x * blockDim.x;
  for (; i < n; i += stride) z[i] = (z[i] > 0.f) ? 1.f : 0.f;
}

__global__ void __launch_bounds__(256) mse_kernel(const float* __restrict__ a,
                                                  const float* __restrict__ b,
                                                  float* __restrict__ out,
                                                  long long n, float inv) {
  __shared__ float red[256];
  const int tid = threadIdx.x;
  float s = 0.f;
  long long i = (long long)blockIdx.x * 256 + tid;
  long long stride = (long long)gridDim.x * 256;
  for (; i < n; i += stride) { float d = a[i] - b[i]; s += d * d; }
  red[tid] = s; __syncthreads();
  for (int o = 128; o > 0; o >>= 1) { if (tid < o) red[tid] += red[tid + o]; __syncthreads(); }
  if (tid == 0) atomicAdd(out, red[0] * inv);
}

// ---------------------------------------------------------------------------
// Host-side orchestration
// ---------------------------------------------------------------------------
struct WsBufs {
  float *lnA, *lnB, *qb, *kvb, *yb, *hb, *sc;
};

struct BlkParams {
  const float *ln1, *ln2;
  const float *qw, *qbi, *kvw, *kvbi, *pw, *pb;
  const float *fcw, *fcb, *mw, *mb;
  const float *lnq, *lnkv;
};

static inline void launch_gemm(hipStream_t st, const float* A, int lda,
                               const float* B, int ldb, int transB,
                               const float* bias, const float* resid, int ldr,
                               float* C, int ldc, int M, int N, int K,
                               float alpha, int act) {
  dim3 g((N + TILE_N - 1) / TILE_N, (M + TILE_M - 1) / TILE_M), b(256);
  const bool aligned = (M % TILE_M == 0) && (N % TILE_N == 0) && (K % TILE_K == 0) &&
                       (lda % 4 == 0) && (!transB || (ldb % 4 == 0));
  if (aligned) {
    if (transB)
      gemm_wmma_kernel<true, true><<<g, b, 0, st>>>(A, lda, B, ldb, bias, resid, ldr, C, ldc, M, N, K, alpha, act);
    else
      gemm_wmma_kernel<true, false><<<g, b, 0, st>>>(A, lda, B, ldb, bias, resid, ldr, C, ldc, M, N, K, alpha, act);
  } else {
    if (transB)
      gemm_wmma_kernel<false, true><<<g, b, 0, st>>>(A, lda, B, ldb, bias, resid, ldr, C, ldc, M, N, K, alpha, act);
    else
      gemm_wmma_kernel<false, false><<<g, b, 0, st>>>(A, lda, B, ldb, bias, resid, ldr, C, ldc, M, N, K, alpha, act);
  }
}

// Multi-head attention: q/kv already projected into w.qb [BT,D], w.kvb [BT,2D].
// Output heads written into w.yb [BT,D] (concat-by-layout).
static void run_attn(hipStream_t st, WsBufs& w, int T, int D, int causal) {
  const int dh = D / NHEAD;
  const float al = 1.0f / sqrtf((float)dh);
  for (int b = 0; b < B_SZ; ++b) {
    for (int h = 0; h < NHEAD; ++h) {
      const float* Q = w.qb + (size_t)b * T * D + h * dh;
      const float* K = w.kvb + (size_t)b * T * 2 * D + h * dh;
      const float* V = K + D;
      // scores = (Q x K^T) / sqrt(dh)   [T,T]
      launch_gemm(st, Q, D, K, 2 * D, 1, nullptr, nullptr, 0, w.sc, T, T, T, dh, al, 0);
      softmax_kernel<<<T, 256, 0, st>>>(w.sc, T, causal);
      // y_head = softmax(scores) x V    [T,dh] -> strided into yb
      launch_gemm(st, w.sc, T, V, 2 * D, 0, nullptr, nullptr, 0,
                  w.yb + (size_t)b * T * D + h * dh, D, T, dh, T, 1.0f, 0);
    }
  }
}

// x = x + mha(ln1(x)); x = x + mlp(ln2(x))
static void run_block(hipStream_t st, WsBufs& w, const BlkParams& p,
                      const float* xin, float* xout, int T, int D, int causal) {
  const int BT = B_SZ * T;
  ln_kernel<<<BT, 256, 0, st>>>(xin, p.ln1, w.lnA, D);
  launch_gemm(st, w.lnA, D, p.qw, D, 0, p.qbi, nullptr, 0, w.qb, D, BT, D, D, 1.0f, 0);
  launch_gemm(st, w.lnA, D, p.kvw, 2 * D, 0, p.kvbi, nullptr, 0, w.kvb, 2 * D, BT, 2 * D, D, 1.0f, 0);
  run_attn(st, w, T, D, causal);
  launch_gemm(st, w.yb, D, p.pw, D, 0, p.pb, xin, D, xout, D, BT, D, D, 1.0f, 0);
  ln_kernel<<<BT, 256, 0, st>>>(xout, p.ln2, w.lnA, D);
  launch_gemm(st, w.lnA, D, p.fcw, 4 * D, 0, p.fcb, nullptr, 0, w.hb, 4 * D, BT, 4 * D, D, 1.0f, 1);
  launch_gemm(st, w.hb, 4 * D, p.mw, D, 0, p.mb, xout, D, xout, D, BT, D, 4 * D, 1.0f, 0);
}

// y = q + mha(lnq(q), lnkv(kv)); y = y + mlp(ln2(y))
static void run_cblock(hipStream_t st, WsBufs& w, const BlkParams& p,
                       const float* kvin, const float* qin, float* yout,
                       int T, int D, int causal) {
  const int BT = B_SZ * T;
  ln_kernel<<<BT, 256, 0, st>>>(qin, p.lnq, w.lnA, D);
  ln_kernel<<<BT, 256, 0, st>>>(kvin, p.lnkv, w.lnB, D);
  launch_gemm(st, w.lnA, D, p.qw, D, 0, p.qbi, nullptr, 0, w.qb, D, BT, D, D, 1.0f, 0);
  launch_gemm(st, w.lnB, D, p.kvw, 2 * D, 0, p.kvbi, nullptr, 0, w.kvb, 2 * D, BT, 2 * D, D, 1.0f, 0);
  run_attn(st, w, T, D, causal);
  launch_gemm(st, w.yb, D, p.pw, D, 0, p.pb, qin, D, yout, D, BT, D, D, 1.0f, 0);
  ln_kernel<<<BT, 256, 0, st>>>(yout, p.ln2, w.lnA, D);
  launch_gemm(st, w.lnA, D, p.fcw, 4 * D, 0, p.fcb, nullptr, 0, w.hb, 4 * D, BT, 4 * D, D, 1.0f, 1);
  launch_gemm(st, w.hb, 4 * D, p.mw, D, 0, p.mb, yout, D, yout, D, BT, D, 4 * D, 1.0f, 0);
}

static BlkParams block_params(void* const* d_in, int i0) {
  auto F = [&](int i) { return (const float*)d_in[i]; };
  BlkParams p{};
  p.ln1 = F(i0 + 0);  p.ln2 = F(i0 + 1);
  p.qw  = F(i0 + 2);  p.qbi = F(i0 + 3);
  p.kvw = F(i0 + 4);  p.kvbi = F(i0 + 5);
  p.pw  = F(i0 + 6);  p.pb  = F(i0 + 7);
  p.fcw = F(i0 + 8);  p.fcb = F(i0 + 9);
  p.mw  = F(i0 + 10); p.mb  = F(i0 + 11);
  return p;
}

static BlkParams cblock_params(void* const* d_in, int i0) {
  auto F = [&](int i) { return (const float*)d_in[i]; };
  BlkParams p{};
  p.ln2 = F(i0 + 0);
  p.qw  = F(i0 + 1);  p.qbi = F(i0 + 2);
  p.kvw = F(i0 + 3);  p.kvbi = F(i0 + 4);
  p.pw  = F(i0 + 5);  p.pb  = F(i0 + 6);
  p.fcw = F(i0 + 7);  p.fcb = F(i0 + 8);
  p.mw  = F(i0 + 9);  p.mb  = F(i0 + 10);
  p.lnq = F(i0 + 11); p.lnkv = F(i0 + 12);
  return p;
}

extern "C" void kernel_launch(void* const* d_in, const int* in_sizes, int n_in,
                              void* d_out, int out_size, void* d_ws, size_t ws_size,
                              hipStream_t stream) {
  (void)in_sizes; (void)n_in; (void)ws_size;
  auto F = [&](int i) { return (const float*)d_in[i]; };

  // ---- Input leaf map (setup_inputs() insertion order) ----
  const float* x = F(0);                          // [B, 2048, 768]
  const BlkParams up_scan     = block_params(d_in, 1);    // 1..12
  const float* up_norm        = F(13);
  const float* ug_fc1w = F(14); const float* ug_fc1b = F(15);
  const float* ug_fc2w = F(16); const float* ug_fc2b = F(17);
  const BlkParams query_block = block_params(d_in, 18);   // 18..29
  const float* upw = F(30); const float* upb_ = F(31);    // up_proj
  const float* encw = F(32); const float* encb = F(33);
  const float* decw = F(34); const float* decb = F(35);
  const float* pew = F(36); const float* peb_ = F(37);    // output_pos_emb
  // 38..41 down_gate, 42 gate_norm: dead downstream -> skipped
  const BlkParams ds1 = cblock_params(d_in, 43);          // down_scatter (D=1024)
  const BlkParams ds2 = cblock_params(d_in, 56);          // down_scatter2 (D=768)
  const float* dpw = F(69); const float* dpb = F(70);     // down_proj
  const BlkParams down_scan = block_params(d_in, 71);     // 71..82

  // ---- Workspace bump allocator ----
  size_t off = 0;
  auto alloc = [&](size_t elems) {
    float* p = (float*)((char*)d_ws + off);
    off += ((elems * sizeof(float) + 255) / 256) * 256;
    return p;
  };
  const size_t BT   = (size_t)B_SZ * OUTER_SEQ;         // 8192
  const size_t BTi  = (size_t)B_SZ * INNER_SEQ;         // 2048
  WsBufs w{};
  w.lnA  = alloc(BT * INNER_DIM);                       // ln scratch (max D)
  w.lnB  = alloc(BT * INNER_DIM);
  w.qb   = alloc(BT * INNER_DIM);
  w.kvb  = alloc(BT * 2 * INNER_DIM);
  w.yb   = alloc(BT * INNER_DIM);
  w.hb   = alloc(BT * 4 * INNER_DIM);                   // MLP hidden (max 4D)
  w.sc   = alloc((size_t)OUTER_SEQ * OUTER_SEQ);        // attention scores
  float* t1   = alloc(BT * INNER_DIM);                  // generic activation buf
  float* t2   = alloc(BT * INNER_DIM);
  float* pdsb = alloc(BT * OUTER_DIM);
  float* pe   = alloc((size_t)OUTER_SEQ * INNER_DIM);
  float* qpj  = alloc((size_t)OUTER_SEQ * OUTER_DIM);
  float* upv  = alloc(BTi * INNER_DIM);                 // "up"/"passed"
  float* aeb  = alloc(BTi * INNER_DIM);
  float* zb   = alloc(BTi * NCODES);
  float* gate = alloc(BT);
  int*   idx  = (int*)alloc((size_t)B_SZ * INNER_SEQ);

  float* out = (float*)d_out;
  float* aux = out + (size_t)(out_size - 1);

  // ================= abstract_up =================
  // scan = block(up_scan, x, causal=False)
  run_block(stream, w, up_scan, x, t1, OUTER_SEQ, OUTER_DIM, 0);
  // scan = ln(scan, up_norm)
  ln_kernel<<<(int)BT, 256, 0, stream>>>(t1, up_norm, t2, OUTER_DIM);
  // gate = fc2(relu(fc1(scan)))[..., 0]
  launch_gemm(stream, t2, OUTER_DIM, ug_fc1w, OUTER_DIM / 4, 0, ug_fc1b, nullptr, 0,
              w.hb, OUTER_DIM / 4, (int)BT, OUTER_DIM / 4, OUTER_DIM, 1.0f, 2);
  launch_gemm(stream, w.hb, OUTER_DIM / 4, ug_fc2w, 1, 0, ug_fc2b, nullptr, 0,
              gate, 1, (int)BT, 1, OUTER_DIM / 4, 1.0f, 0);
  // gathered = block(query_block, x, causal=True)
  run_block(stream, w, query_block, x, t1, OUTER_SEQ, OUTER_DIM, 1);
  // idx = top_k(gate, 512); gathered = gather(gathered, idx)
  {
    dim3 g(OUTER_SEQ / 256, B_SZ);
    topk_kernel<<<g, 256, 0, stream>>>(gate, idx, OUTER_SEQ, INNER_SEQ);
  }
  float* gth = t2;  // [B, 512, 768]
  gather_kernel<<<B_SZ * INNER_SEQ, 256, 0, stream>>>(t1, idx, gth, OUTER_SEQ, INNER_SEQ, OUTER_DIM);
  // up = up_proj(gathered)  [B, 512, 1024]
  launch_gemm(stream, gth, OUTER_DIM, upw, INNER_DIM, 0, upb_, nullptr, 0,
              upv, INNER_DIM, (int)BTi, INNER_DIM, OUTER_DIM, 1.0f, 0);
  // Autoencoder aux: z = enc(up); hard = z>0; ae_up = dec(hard); aux = mean((up-ae_up)^2)
  launch_gemm(stream, upv, INNER_DIM, encw, NCODES, 0, encb, nullptr, 0,
              zb, NCODES, (int)BTi, NCODES, INNER_DIM, 1.0f, 0);
  binarize_kernel<<<256, 256, 0, stream>>>(zb, (long long)BTi * NCODES);
  launch_gemm(stream, zb, NCODES, decw, INNER_DIM, 0, decb, nullptr, 0,
              aeb, INNER_DIM, (int)BTi, INNER_DIM, NCODES, 1.0f, 0);
  fill_kernel<<<1, 32, 0, stream>>>(aux, 0.0f, 1);
  mse_kernel<<<256, 256, 0, stream>>>(upv, aeb, aux, (long long)BTi * INNER_DIM,
                                      1.0f / (float)(BTi * INNER_DIM));

  // ================= abstract_down =================
  // (down_gate / gate_norm / down_scan(p_down) subgraph is dead -> eliminated)
  // pos_emb = eye @ W + b = rows of W + bias
  rowbias_kernel<<<1024, 256, 0, stream>>>(pew, peb_, pe,
                                           (long long)OUTER_SEQ * INNER_DIM, INNER_DIM);
  // scattered = zeros; scattered[b, idx] = passed
  fill_kernel<<<2048, 256, 0, stream>>>(t1, 0.0f, (long long)BT * INNER_DIM);
  scatter_kernel<<<B_SZ * INNER_SEQ, 256, 0, stream>>>(upv, idx, t1, OUTER_SEQ, INNER_SEQ, INNER_DIM);
  // q = broadcast(pos_emb) to [B, 2048, 1024]
  bcast_add_kernel<<<2048, 256, 0, stream>>>(pe, nullptr, t2, (long long)BT * INNER_DIM,
                                             (long long)OUTER_SEQ * INNER_DIM, 0);
  // pds = cblock(down_scatter, kv=scattered, q=pos_emb, causal)
  run_cblock(stream, w, ds1, t1, t2, t1, OUTER_SEQ, INNER_DIM, 1);
  // pds = down_proj(pds)  [B, 2048, 768]
  launch_gemm(stream, t1, INNER_DIM, dpw, OUTER_DIM, 0, dpb, nullptr, 0,
              pdsb, OUTER_DIM, (int)BT, OUTER_DIM, INNER_DIM, 1.0f, 0);
  // query = down_proj(pos_emb) + x
  launch_gemm(stream, pe, INNER_DIM, dpw, OUTER_DIM, 0, dpb, nullptr, 0,
              qpj, OUTER_DIM, OUTER_SEQ, OUTER_DIM, INNER_DIM, 1.0f, 0);
  bcast_add_kernel<<<2048, 256, 0, stream>>>(qpj, x, t2, (long long)BT * OUTER_DIM,
                                             (long long)OUTER_SEQ * OUTER_DIM, 1);
  // out = cblock(down_scatter2, kv=pds, q=query, causal)
  run_cblock(stream, w, ds2, pdsb, t2, t1, OUTER_SEQ, OUTER_DIM, 1);
  // out = block(down_scan, out, causal) -> final output
  run_block(stream, w, down_scan, t1, out, OUTER_SEQ, OUTER_DIM, 1);
}
